// G_22093311770918
// MI455X (gfx1250) — compile-verified, matrix operand order checked by reading, not run on hardware
//
#include <hip/hip_runtime.h>
#include <hip/hip_bf16.h>

// Problem dims (fixed by the reference)
#define HDIM 1024
#define BATCH 128
#define NCLS 256

typedef __attribute__((ext_vector_type(16))) __bf16 v16bf;
typedef __attribute__((ext_vector_type(8)))  float  v8f;
typedef __attribute__((ext_vector_type(4)))  float  v4f;

// ---------------------------------------------------------------------------
// Tile pipeline: load_raw issues four independent global_load_b128 (fp32),
// cvt_tile packs to the bf16 WMMA operand layout. The K-loop software-
// pipelines: loads for strip k+32 are issued before strip k's convert/WMMA,
// so a full strip (32 loads) stays in flight.
//
// CDNA5 16-bit A-operand layout (CDNA5 ISA §7.12.2), wave32:
//   lanes 0-15  : row M = lane,    K = {0..7, 16..23}
//   lanes 16-31 : row M = lane-16, K = {8..15, 24..31}
// B (KxN, col n per lane) loads identically from the row-major [N,K] weights.
// ---------------------------------------------------------------------------
struct RawTile { v4f f0, f1, f2, f3; };

__device__ __forceinline__ RawTile load_raw(const float* __restrict__ p) {
  const v4f* p4 = reinterpret_cast<const v4f*>(p);   // 32B-aligned
  RawTile t;
  t.f0 = p4[0];   // K + {0..3}
  t.f1 = p4[1];   // K + {4..7}
  t.f2 = p4[4];   // K + {16..19}
  t.f3 = p4[5];   // K + {20..23}
  return t;
}

__device__ __forceinline__ v16bf cvt_tile(const RawTile& t) {
  v16bf r;
#pragma unroll
  for (int i = 0; i < 4; ++i) {
    r[i]      = (__bf16)t.f0[i];
    r[4 + i]  = (__bf16)t.f1[i];
    r[8 + i]  = (__bf16)t.f2[i];
    r[12 + i] = (__bf16)t.f3[i];
  }
  return r;
}

__device__ __forceinline__ v8f wmma_bf16(v16bf a, v16bf b, v8f c) {
  // 8 args: (neg_a, A, neg_b, B, c_mod, C, reuse_a, reuse_b)
  return __builtin_amdgcn_wmma_f32_16x16x32_bf16(false, a, false, b,
                                                 (short)0, c, false, false);
}

__device__ __forceinline__ float sigmoidf_fast(float v) {
  return 1.0f / (1.0f + __expf(-v));
}

// Per-lane base pointer for a tile whose first row is `row` in a row-major
// [_, ld] fp32 matrix (lane-dependent half-K offset baked in).
__device__ __forceinline__ const float* tile_base(const float* __restrict__ mat,
                                                  int ld, int row, int lane) {
  return mat + (size_t)(row + (lane & 15)) * (size_t)ld + (size_t)((lane >> 4) << 3);
}

// ---------------------------------------------------------------------------
// Kernel 1: fused GRU cell.
// grid = 64 blocks (one 16-col slice of H each), block = 256 threads = 8 waves.
// Wave w owns batch rows [16w, 16w+16). Each wave accumulates six 16x16 tiles
// (i_r,i_z,i_n with A=x; h_r,h_z,h_n with A=h1), applies the gate math
// in-register and emits h1' and res = h1' + x.
// ---------------------------------------------------------------------------
__global__ __launch_bounds__(256, 1)
void k_gru_fused(const float* __restrict__ x, const float* __restrict__ h1,
                 const float* __restrict__ w_ih, const float* __restrict__ w_hh,
                 const float* __restrict__ b_ih, const float* __restrict__ b_hh,
                 float* __restrict__ h1p, float* __restrict__ res) {
  const int lane = threadIdx.x & 31;
  const int wave = threadIdx.x >> 5;
  const int m0   = wave * 16;
  const int n0   = blockIdx.x * 16;

  // Hoisted per-lane tile base pointers (K advances by +32 floats per iter).
  const float* px  = tile_base(x,    HDIM, m0,            lane);
  const float* ph  = tile_base(h1,   HDIM, m0,            lane);
  const float* pir = tile_base(w_ih, HDIM, n0,            lane);
  const float* piz = tile_base(w_ih, HDIM, n0 + HDIM,     lane);
  const float* pin = tile_base(w_ih, HDIM, n0 + 2 * HDIM, lane);
  const float* phr = tile_base(w_hh, HDIM, n0,            lane);
  const float* phz = tile_base(w_hh, HDIM, n0 + HDIM,     lane);
  const float* phn = tile_base(w_hh, HDIM, n0 + 2 * HDIM, lane);

  v8f a_ir = {}, a_iz = {}, a_in = {}, a_hr = {}, a_hz = {}, a_hn = {};

  // Software pipeline prologue: strip k=0 in flight.
  RawTile cx = load_raw(px),  ch = load_raw(ph);
  RawTile c0 = load_raw(pir), c1 = load_raw(piz), c2 = load_raw(pin);
  RawTile c3 = load_raw(phr), c4 = load_raw(phz), c5 = load_raw(phn);

#pragma unroll 2
  for (int k0 = 32; k0 < HDIM; k0 += 32) {
    // Issue next strip's 32 loads (independent of current compute).
    RawTile nx = load_raw(px  + k0), nh = load_raw(ph  + k0);
    RawTile n0_ = load_raw(pir + k0), n1 = load_raw(piz + k0);
    RawTile n2 = load_raw(pin + k0), n3 = load_raw(phr + k0);
    RawTile n4 = load_raw(phz + k0), n5 = load_raw(phn + k0);

    // Convert + matrix ops on the current strip.
    v16bf ax = cvt_tile(cx);
    v16bf ah = cvt_tile(ch);
    a_ir = wmma_bf16(ax, cvt_tile(c0), a_ir);
    a_iz = wmma_bf16(ax, cvt_tile(c1), a_iz);
    a_in = wmma_bf16(ax, cvt_tile(c2), a_in);
    a_hr = wmma_bf16(ah, cvt_tile(c3), a_hr);
    a_hz = wmma_bf16(ah, cvt_tile(c4), a_hz);
    a_hn = wmma_bf16(ah, cvt_tile(c5), a_hn);

    // Rotate buffers (renamed away under unrolling).
    cx = nx; ch = nh; c0 = n0_; c1 = n1; c2 = n2; c3 = n3; c4 = n4; c5 = n5;
  }

  // Pipeline epilogue: last strip.
  {
    v16bf ax = cvt_tile(cx);
    v16bf ah = cvt_tile(ch);
    a_ir = wmma_bf16(ax, cvt_tile(c0), a_ir);
    a_iz = wmma_bf16(ax, cvt_tile(c1), a_iz);
    a_in = wmma_bf16(ax, cvt_tile(c2), a_in);
    a_hr = wmma_bf16(ah, cvt_tile(c3), a_hr);
    a_hz = wmma_bf16(ah, cvt_tile(c4), a_hz);
    a_hn = wmma_bf16(ah, cvt_tile(c5), a_hn);
  }

  // Gate epilogue. C/D layout: element (m,n) -> vgpr m&7, lane n + 16*(m>>3).
  const int n = n0 + (lane & 15);
  const float bir  = b_ih[n],            bhr = b_hh[n];
  const float biz  = b_ih[n + HDIM],     bhz = b_hh[n + HDIM];
  const float bin_ = b_ih[n + 2 * HDIM], bhn = b_hh[n + 2 * HDIM];

#pragma unroll
  for (int v = 0; v < 8; ++v) {
    const int m = m0 + v + ((lane >> 4) << 3);
    const size_t idx = (size_t)m * HDIM + n;
    const float r  = sigmoidf_fast((a_ir[v] + bir) + (a_hr[v] + bhr));
    const float z  = sigmoidf_fast((a_iz[v] + biz) + (a_hz[v] + bhz));
    const float ng = tanhf((a_in[v] + bin_) + r * (a_hn[v] + bhn));
    const float hp = (1.0f - z) * ng + z * h1[idx];
    h1p[idx] = hp;           // h1' output
    res[idx] = hp + x[idx];  // residual into fc1
  }
}

// ---------------------------------------------------------------------------
// Kernels 2/3: C = act(A @ W^T + bias). One 16-col N-tile per block, one
// 16-row M-tile per wave (8 waves = full batch of 128). Same software
// pipeline: next K-strip's 8 loads issue before current strip's WMMA.
// ---------------------------------------------------------------------------
template <bool RELU>
__global__ __launch_bounds__(256, 1)
void k_gemm_bias(const float* __restrict__ A, const float* __restrict__ W,
                 const float* __restrict__ bias, float* __restrict__ out,
                 int ncols /* output row stride */) {
  const int lane = threadIdx.x & 31;
  const int wave = threadIdx.x >> 5;
  const int m0   = wave * 16;
  const int n0   = blockIdx.x * 16;

  const float* pa = tile_base(A, HDIM, m0, lane);
  const float* pw = tile_base(W, HDIM, n0, lane);

  v8f acc = {};
  RawTile ca = load_raw(pa), cw = load_raw(pw);

#pragma unroll 4
  for (int k0 = 32; k0 < HDIM; k0 += 32) {
    RawTile na = load_raw(pa + k0);
    RawTile nw = load_raw(pw + k0);
    acc = wmma_bf16(cvt_tile(ca), cvt_tile(cw), acc);
    ca = na; cw = nw;
  }
  acc = wmma_bf16(cvt_tile(ca), cvt_tile(cw), acc);

  const int n = n0 + (lane & 15);
  const float bn = bias[n];
#pragma unroll
  for (int v = 0; v < 8; ++v) {
    const int m = m0 + v + ((lane >> 4) << 3);
    float val = acc[v] + bn;
    if (RELU) val = fmaxf(val, 0.0f);
    out[(size_t)m * ncols + n] = val;
  }
}

extern "C" void kernel_launch(void* const* d_in, const int* in_sizes, int n_in,
                              void* d_out, int out_size, void* d_ws, size_t ws_size,
                              hipStream_t stream) {
  const float* x     = (const float*)d_in[0];
  const float* h1    = (const float*)d_in[1];
  const float* w_ih  = (const float*)d_in[2];
  const float* w_hh  = (const float*)d_in[3];
  const float* b_ih  = (const float*)d_in[4];
  const float* b_hh  = (const float*)d_in[5];
  const float* w_fc1 = (const float*)d_in[6];
  const float* b_fc1 = (const float*)d_in[7];
  const float* w_fc3 = (const float*)d_in[8];
  const float* b_fc3 = (const float*)d_in[9];

  // d_out = [out (128x256) | h1_prime (128x1024)] flat fp32
  float* out = (float*)d_out;
  float* h1p = out + (size_t)BATCH * NCLS;

  // workspace: res (128x1024) | relu1 (128x1024)
  float* res   = (float*)d_ws;
  float* relu1 = res + (size_t)BATCH * HDIM;

  // Kernel 1: GRU cell -> h1', res
  k_gru_fused<<<HDIM / 16, 256, 0, stream>>>(x, h1, w_ih, w_hh, b_ih, b_hh,
                                             h1p, res);
  // Kernel 2: relu1 = relu(res @ w_fc1^T + b_fc1)
  k_gemm_bias<true><<<HDIM / 16, 256, 0, stream>>>(res, w_fc1, b_fc1, relu1, HDIM);
  // Kernel 3: out = relu1 @ w_fc3^T + b_fc3
  k_gemm_bias<false><<<NCLS / 16, 256, 0, stream>>>(relu1, w_fc3, b_fc3, out, NCLS);
}
